// Encoder_37117107372137
// MI455X (gfx1250) — compile-verified, matrix operand order checked by reading, not run on hardware
//
#include <hip/hip_runtime.h>

typedef __attribute__((ext_vector_type(16))) __bf16 v16bf;
typedef __attribute__((ext_vector_type(8)))  __bf16 v8bf;
typedef __attribute__((ext_vector_type(8)))  float  v8f;

__device__ __forceinline__ __bf16 f2bf(float f) {
  union { float f; unsigned u; } v; v.f = f;
  unsigned r = v.u + 0x7FFFu + ((v.u >> 16) & 1u);   // round-to-nearest-even
  union { unsigned short s; __bf16 b; } o;
  o.s = (unsigned short)(r >> 16);
  return o.b;
}

// ---------------------------------------------------------------------------
// C[M,N] = A[M,K] x B[K,N], A given row-major bf16, B given TRANSPOSED
// (Bt[N][K] bf16) so each lane's K-run is contiguous -> b128 loads.
// One wave = one 16x16 C tile; blockDim = 32*(N/16); K compile-time.
// Lane layout (ISA 7.12.2): lane l row/col = l&15; lanes 0-15 hold
// K {0..7,16..23}, lanes 16-31 hold K {8..15,24..31} per 32-K step.
// ---------------------------------------------------------------------------
template <int K>
__global__ void gemm_wmma_bf16(const __bf16* __restrict__ A,
                               const __bf16* __restrict__ Bt,
                               float* __restrict__ C, int ldc) {
  const int wave = threadIdx.x >> 5;
  const int lane = threadIdx.x & 31;
  const int m16  = lane & 15;
  const size_t arow = (size_t)(blockIdx.x * 16 + m16) * K;
  const size_t brow = (size_t)(wave * 16 + m16) * K;
  const int khalf = (lane >> 4) << 3;   // 0 or 8

  v8f acc = {};
#pragma unroll
  for (int k0 = 0; k0 < K; k0 += 32) {
    const v8bf alo = *(const v8bf*)(A  + arow + k0 + khalf);
    const v8bf ahi = *(const v8bf*)(A  + arow + k0 + 16 + khalf);
    const v8bf blo = *(const v8bf*)(Bt + brow + k0 + khalf);
    const v8bf bhi = *(const v8bf*)(Bt + brow + k0 + 16 + khalf);
    const v16bf a = __builtin_shufflevector(alo, ahi, 0,1,2,3,4,5,6,7,8,9,10,11,12,13,14,15);
    const v16bf b = __builtin_shufflevector(blo, bhi, 0,1,2,3,4,5,6,7,8,9,10,11,12,13,14,15);
    acc = __builtin_amdgcn_wmma_f32_16x16x32_bf16(
        false, a, false, b, (short)0, acc, false, false);
  }

  const int col  = wave * 16 + m16;
  const int crow = blockIdx.x * 16 + ((lane >> 4) << 3);
#pragma unroll
  for (int r = 0; r < 8; ++r)
    C[(size_t)(crow + r) * ldc + col] = acc[r];
}

// ---------------------------------------------------------------------------
// Precision-conversion helpers (one-shot, bandwidth-trivial)
// ---------------------------------------------------------------------------
__global__ void cvt_bf16_k(const float* __restrict__ in, __bf16* __restrict__ out,
                           long long n) {
  long long i = (long long)blockIdx.x * blockDim.x + threadIdx.x;
  if (i < n) out[i] = f2bf(in[i]);
}

// W[K][N] fp32 -> Wt[N][K] bf16
__global__ void cvt_tr_k(const float* __restrict__ W, __bf16* __restrict__ Wt,
                         int K, int N) {
  int i = blockIdx.x * blockDim.x + threadIdx.x;
  if (i >= K * N) return;
  int k = i / N, n = i % N;
  Wt[(size_t)n * K + k] = f2bf(W[i]);
}

// ---------------------------------------------------------------------------
// Graph aggregation (edge-parallel scatter-add; L2-resident atomics)
// ---------------------------------------------------------------------------
__global__ void init_deg_k(float* __restrict__ deg, int n) {
  int i = blockIdx.x * blockDim.x + threadIdx.x;
  if (i < n) deg[i] = 1.0f;                        // self loop
}

__global__ void edge_deg_k(const int* __restrict__ dst, float* __restrict__ deg, int e) {
  int i = blockIdx.x * blockDim.x + threadIdx.x;
  if (i < e) atomicAdd(&deg[dst[i]], 1.0f);
}

__global__ void rsqrt_k(float* __restrict__ d, int n) {
  int i = blockIdx.x * blockDim.x + threadIdx.x;
  if (i < n) { float v = d[i]; d[i] = v > 0.0f ? rsqrtf(v) : 0.0f; }
}

// out[n,f] = hlin[n,f] * dinv[n]^2 + bias[f]   (self-loop term + bias)
__global__ void init_agg_k(const float* __restrict__ hlin, const float* __restrict__ dinv,
                           const float* __restrict__ bias, float* __restrict__ out,
                           long long total, int F) {
  long long i = (long long)blockIdx.x * blockDim.x + threadIdx.x;
  if (i >= total) return;
  int n = (int)(i / F), f = (int)(i % F);
  float di = dinv[n];
  out[i] = hlin[i] * di * di + bias[f];
}

__global__ void edge_agg_k(const int* __restrict__ src, const int* __restrict__ dst,
                           const float* __restrict__ dinv, const float* __restrict__ hlin,
                           float* __restrict__ out, int E, int F4, int F) {
  long long i = (long long)blockIdx.x * blockDim.x + threadIdx.x;
  if (i >= (long long)E * F4) return;
  int e = (int)(i / F4);
  int f = (int)(i % F4) * 4;
  int s = src[e], d = dst[e];
  float nrm = dinv[s] * dinv[d];
  const float4 hv = *(const float4*)(hlin + (size_t)s * F + f);
  float* op = out + (size_t)d * F + f;
  atomicAdd(op + 0, hv.x * nrm);
  atomicAdd(op + 1, hv.y * nrm);
  atomicAdd(op + 2, hv.z * nrm);
  atomicAdd(op + 3, hv.w * nrm);
}

__global__ void relu_k(float* __restrict__ p, long long n) {
  long long i = (long long)blockIdx.x * blockDim.x + threadIdx.x;
  if (i < n) p[i] = fmaxf(p[i], 0.0f);
}

// ---------------------------------------------------------------------------
extern "C" void kernel_launch(void* const* d_in, const int* in_sizes, int n_in,
                              void* d_out, int out_size, void* d_ws, size_t ws_size,
                              hipStream_t stream) {
  const float* x   = (const float*)d_in[0];
  const int*   ei  = (const int*)d_in[1];
  const int*   yei = (const int*)d_in[2];
  const float* W1  = (const float*)d_in[3];
  const float* b1  = (const float*)d_in[4];
  const float* Wmu = (const float*)d_in[5];
  const float* bmu = (const float*)d_in[6];
  const float* Wls = (const float*)d_in[7];
  const float* bls = (const float*)d_in[8];

  const int DIN = 128, DH = 128, DZ = 32;
  const int N  = in_sizes[0] / DIN;       // 50000 (multiple of 16)
  const int E1 = in_sizes[1] / 2;         // 600000
  const int E2 = in_sizes[2] / 2;         // 200000

  const int* src1 = ei;   const int* dst1 = ei  + E1;
  const int* src2 = yei;  const int* dst2 = yei + E2;

  // ---- workspace layout (all offsets 16B-aligned) ----
  char* wsb = (char*)d_ws;
  float*  dinv = (float*)wsb;                                   // N f32
  __bf16* w1t  = (__bf16*)(wsb + (size_t)65536 * 4);            // 128x128 bf16
  __bf16* wmut = w1t + 128 * 128;                               // 32x128 bf16
  __bf16* wlst = wmut + 32 * 128;                               // 32x128 bf16
  __bf16* abf  = (__bf16*)(wsb + (size_t)(65536 + 16384) * 4);  // N*128 bf16 (x, then h)
  float*  buf1 = (float*)(wsb + (size_t)(65536 + 16384) * 4 + (size_t)N * DH * 2);
  float*  buf2 = buf1 + (size_t)N * DH;

  const int T = 256;
  auto blocks = [](long long total, int t) { return (unsigned)((total + t - 1) / t); };

  // ---- degrees (with self loops) -> dinv ----
  init_deg_k<<<blocks(N, T), T, 0, stream>>>(dinv, N);
  edge_deg_k<<<blocks(E1, T), T, 0, stream>>>(dst1, dinv, E1);
  edge_deg_k<<<blocks(E2, T), T, 0, stream>>>(dst2, dinv, E2);
  rsqrt_k<<<blocks(N, T), T, 0, stream>>>(dinv, N);

  // ---- layer 1: h = relu(agg(x@W1) + b1) ----
  cvt_bf16_k<<<blocks((long long)N * DIN, T), T, 0, stream>>>(x, abf, (long long)N * DIN);
  cvt_tr_k<<<blocks(DIN * DH, T), T, 0, stream>>>(W1, w1t, DIN, DH);
  gemm_wmma_bf16<128><<<N / 16, 32 * (DH / 16), 0, stream>>>(abf, w1t, buf1, DH);

  init_agg_k<<<blocks((long long)N * DH, T), T, 0, stream>>>(buf1, dinv, b1, buf2,
                                                             (long long)N * DH, DH);
  edge_agg_k<<<blocks((long long)E1 * (DH / 4), T), T, 0, stream>>>(src1, dst1, dinv,
                                                                    buf1, buf2, E1, DH / 4, DH);
  edge_agg_k<<<blocks((long long)E2 * (DH / 4), T), T, 0, stream>>>(src2, dst2, dinv,
                                                                    buf1, buf2, E2, DH / 4, DH);
  relu_k<<<blocks((long long)N * DH, T), T, 0, stream>>>(buf2, (long long)N * DH);

  // ---- layer 2: z_mu / z_logstd ----
  cvt_bf16_k<<<blocks((long long)N * DH, T), T, 0, stream>>>(buf2, abf, (long long)N * DH);
  cvt_tr_k<<<blocks(DH * DZ, T), T, 0, stream>>>(Wmu, wmut, DH, DZ);
  cvt_tr_k<<<blocks(DH * DZ, T), T, 0, stream>>>(Wls, wlst, DH, DZ);

  float* zmu_lin = buf1;
  float* zls_lin = buf1 + (size_t)N * DZ;
  gemm_wmma_bf16<128><<<N / 16, 32 * (DZ / 16), 0, stream>>>(abf, wmut, zmu_lin, DZ);
  gemm_wmma_bf16<128><<<N / 16, 32 * (DZ / 16), 0, stream>>>(abf, wlst, zls_lin, DZ);

  float* out_mu = (float*)d_out;                 // (z_mu, z_logstd) concat flat
  float* out_ls = out_mu + (size_t)N * DZ;

  init_agg_k<<<blocks((long long)N * DZ, T), T, 0, stream>>>(zmu_lin, dinv, bmu, out_mu,
                                                             (long long)N * DZ, DZ);
  init_agg_k<<<blocks((long long)N * DZ, T), T, 0, stream>>>(zls_lin, dinv, bls, out_ls,
                                                             (long long)N * DZ, DZ);

  edge_agg_k<<<blocks((long long)E1 * (DZ / 4), T), T, 0, stream>>>(src1, dst1, dinv,
                                                                    zmu_lin, out_mu, E1, DZ / 4, DZ);
  edge_agg_k<<<blocks((long long)E2 * (DZ / 4), T), T, 0, stream>>>(src2, dst2, dinv,
                                                                    zmu_lin, out_mu, E2, DZ / 4, DZ);
  edge_agg_k<<<blocks((long long)E1 * (DZ / 4), T), T, 0, stream>>>(src1, dst1, dinv,
                                                                    zls_lin, out_ls, E1, DZ / 4, DZ);
  edge_agg_k<<<blocks((long long)E2 * (DZ / 4), T), T, 0, stream>>>(src2, dst2, dinv,
                                                                    zls_lin, out_ls, E2, DZ / 4, DZ);
}
